// self_attention_88905823027932
// MI455X (gfx1250) — matrix operand order repaired and checked by reading, not compiled, not run on hardware
//
#include <hip/hip_runtime.h>
#include <stdint.h>

// Self-attention N=4096, D=1024, fp32 in/out on MI455X (gfx1250).
// Compute-bound (~378 GFLOP vs ~21us HBM floor) -> all matmuls via
// v_wmma_f32_16x16x32_bf16 (bf16 operands, fp32 accumulate).
// GEMMs consume B as B^T row-major so A and B WMMA fragments are contiguous
// 16B LDS loads. Staging uses gfx1250 async global->LDS DMA (ASYNCcnt) with
// double-buffered LDS so HBM latency overlaps WMMA of the previous tile.

#define NN 4096
#define DD 1024

#define USE_ASYNC 1   // global_load_async_to_lds_b128 staging path

typedef __attribute__((ext_vector_type(16))) __bf16 v16bf;
typedef __attribute__((ext_vector_type(8)))  float  v8f;

__device__ __forceinline__ unsigned short f2bf(float f) {
    unsigned u = __float_as_uint(f);
    unsigned r = u + 0x7fffu + ((u >> 16) & 1u);   // round-to-nearest-even
    return (unsigned short)(r >> 16);
}

// ---------------------------------------------------------------- conversion
__global__ void cvt_f32_bf16(const float* __restrict__ in,
                             unsigned short* __restrict__ out, int n) {
    for (int i = blockIdx.x * blockDim.x + threadIdx.x; i < n;
         i += gridDim.x * blockDim.x)
        out[i] = f2bf(in[i]);
}

// out[c][r] = (bf16) in[r][c]   (in: rows x cols fp32)
__global__ void transpose_f32_to_bf16(const float* __restrict__ in,
                                      unsigned short* __restrict__ out,
                                      int rows, int cols) {
    __shared__ float tile[32][33];
    int tx = threadIdx.x, ty = threadIdx.y;
    int x = blockIdx.x * 32 + tx;
#pragma unroll
    for (int j = 0; j < 32; j += 8)
        tile[ty + j][tx] = in[(size_t)(blockIdx.y * 32 + ty + j) * cols + x];
    __syncthreads();
    int xo = blockIdx.y * 32 + tx;
#pragma unroll
    for (int j = 0; j < 32; j += 8)
        out[(size_t)(blockIdx.x * 32 + ty + j) * rows + xo] =
            f2bf(tile[tx][ty + j]);
}

// out[c][r] = in[r][c]   (bf16 -> bf16)
__global__ void transpose_bf16(const unsigned short* __restrict__ in,
                               unsigned short* __restrict__ out,
                               int rows, int cols) {
    __shared__ unsigned short tile[32][33];
    int tx = threadIdx.x, ty = threadIdx.y;
    int x = blockIdx.x * 32 + tx;
#pragma unroll
    for (int j = 0; j < 32; j += 8)
        tile[ty + j][tx] = in[(size_t)(blockIdx.y * 32 + ty + j) * cols + x];
    __syncthreads();
    int xo = blockIdx.y * 32 + tx;
#pragma unroll
    for (int j = 0; j < 32; j += 8)
        out[(size_t)(blockIdx.x * 32 + ty + j) * rows + xo] = tile[tx][ty + j];
}

// ---------------------------------------------------------------- GEMM
// C(MxN, fp32 acc) = A(MxK bf16 row-major) * Bt(NxK bf16 row-major)^T
// Block tile 256x128, BK=64, 256 threads = 8 waves (4 M x 2 N),
// wave tile 64x64 = 4x4 WMMA accumulators (16 wmma / 16 frag-loads per BK/2).
#define BM 256
#define BN 128
#define BK 64
#define LDSS 72               // BK+8 pad: 144B row stride = 36 banks, cycle 16
#define ASZ (BM * LDSS)       // elems per A buffer
#define BSZ (BN * LDSS)

union FragAB { v16bf v; uint4 q[2]; };

__device__ __forceinline__ void async_cp_b128(const unsigned short* g,
                                              const unsigned short* l) {
#if USE_ASYNC
    unsigned lds = (unsigned)(uintptr_t)(const void*)l;   // wave LDS offset
    unsigned long long ga = (unsigned long long)(uintptr_t)g;
    asm volatile("global_load_async_to_lds_b128 %0, %1, off"
                 :: "v"(lds), "v"(ga) : "memory");
#endif
}
__device__ __forceinline__ void wait_async0() {
#if USE_ASYNC
    asm volatile("s_wait_asynccnt 0x0" ::: "memory");
#endif
}

template <bool F32OUT>
__global__ __launch_bounds__(256) void
gemm_bt_bf16(const unsigned short* __restrict__ A,
             const unsigned short* __restrict__ Bt,
             void* __restrict__ Cout, int M, int Ncols, int K) {
    __shared__ __align__(16) unsigned short sA[2 * ASZ];
    __shared__ __align__(16) unsigned short sB[2 * BSZ];

    const int tid  = threadIdx.x;
    const int lane = tid & 31;
    const int wave = tid >> 5;
    const int half = lane >> 4;      // 0: lanes 0-15, 1: lanes 16-31
    const int lr   = lane & 15;
    const int waveM = wave & 3;      // 4 waves x 64 rows
    const int waveN = wave >> 2;     // 2 waves x 64 cols
    const int blockRow = blockIdx.y * BM;
    const int blockCol = blockIdx.x * BN;

    v8f zero = {};
    v8f acc[4][4];
#pragma unroll
    for (int i = 0; i < 4; ++i)
#pragma unroll
        for (int j = 0; j < 4; ++j) acc[i][j] = zero;

    // Issue one LDS buffer's worth of async b128 copies for K-tile k0.
    auto prefetch = [&](int buf, int k0) {
#if USE_ASYNC
#pragma unroll
        for (int i = 0; i < 8; ++i) {               // A: 256x64 = 2048 chunks
            int c = i * 256 + tid;
            int row = c >> 2, off = (c & 3) * 8;
            async_cp_b128(A + (size_t)(blockRow + row) * K + k0 + off,
                          sA + buf * ASZ + row * LDSS + off);
        }
#pragma unroll
        for (int i = 0; i < 4; ++i) {               // B: 128x64 = 1024 chunks
            int c = i * 256 + tid;
            int row = c >> 2, off = (c & 3) * 8;
            async_cp_b128(Bt + (size_t)(blockCol + row) * K + k0 + off,
                          sB + buf * BSZ + row * LDSS + off);
        }
#else
        // synchronous fallback: global -> VGPR -> LDS
        const int r = tid >> 1, cc = (tid & 1) * 32;
#pragma unroll
        for (int rr = 0; rr < 2; ++rr) {            // A: 256 rows, 2 per thread
            const uint4* ga = (const uint4*)(A + (size_t)(blockRow + r + rr * 128) * K + k0 + cc);
            uint4* la = (uint4*)(sA + buf * ASZ + (r + rr * 128) * LDSS + cc);
            la[0] = ga[0]; la[1] = ga[1]; la[2] = ga[2]; la[3] = ga[3];
        }
        const uint4* gb = (const uint4*)(Bt + (size_t)(blockCol + r) * K + k0 + cc);
        uint4* lb = (uint4*)(sB + buf * BSZ + r * LDSS + cc);
        lb[0] = gb[0]; lb[1] = gb[1]; lb[2] = gb[2]; lb[3] = gb[3];
#endif
    };

    const int nk = K / BK;
    prefetch(0, 0);

    for (int kt = 0; kt < nk; ++kt) {
        const int cur = kt & 1;
        wait_async0();          // my async copies into buf `cur` are done
        __syncthreads();        // everyone's are; last tile's readers passed
        if (kt + 1 < nk) prefetch(1 - cur, (kt + 1) * BK);
#if !USE_ASYNC
        __syncthreads();
#endif
        const unsigned short* a0 = sA + cur * ASZ;
        const unsigned short* b0 = sB + cur * BSZ;
#pragma unroll
        for (int ks = 0; ks < BK; ks += 32) {
            FragAB a[4], b[4];
            // A frag: lane = row (M), K = half*8 + {0..7, 16..23}
#pragma unroll
            for (int fm = 0; fm < 4; ++fm) {
                int base = (waveM * 64 + fm * 16 + lr) * LDSS + ks + half * 8;
                a[fm].q[0] = *(const uint4*)(a0 + base);
                a[fm].q[1] = *(const uint4*)(a0 + base + 16);
            }
            // B frag: lane = col (N) = row of Bt, K = half*8 + {0..7, 16..23}
#pragma unroll
            for (int fn = 0; fn < 4; ++fn) {
                int base = (waveN * 64 + fn * 16 + lr) * LDSS + ks + half * 8;
                b[fn].q[0] = *(const uint4*)(b0 + base);
                b[fn].q[1] = *(const uint4*)(b0 + base + 16);
            }
#pragma unroll
            for (int fm = 0; fm < 4; ++fm)
#pragma unroll
                for (int fn = 0; fn < 4; ++fn)
                    acc[fm][fn] = __builtin_amdgcn_wmma_f32_16x16x32_bf16(
                        false, a[fm].v, false, b[fn].v, (short)0, acc[fm][fn],
                        false, false);
        }
#if !USE_ASYNC
        __syncthreads();
#endif
    }

    // Epilogue: C VGPR rr -> row rr (lanes 0-15) / rr+8 (lanes 16-31)
#pragma unroll
    for (int fm = 0; fm < 4; ++fm) {
        int row0 = blockRow + waveM * 64 + fm * 16 + half * 8;
#pragma unroll
        for (int fn = 0; fn < 4; ++fn) {
            int col = blockCol + waveN * 64 + fn * 16 + lr;
            union { v8f v; float f[8]; } u;
            u.v = acc[fm][fn];
#pragma unroll
            for (int rr = 0; rr < 8; ++rr) {
                size_t idx = (size_t)(row0 + rr) * Ncols + col;
                if (F32OUT)
                    ((float*)Cout)[idx] = u.f[rr];
                else
                    ((unsigned short*)Cout)[idx] = f2bf(u.f[rr]);
            }
        }
    }
}

// ---------------------------------------------------------------- softmax
__global__ __launch_bounds__(256) void
softmax_rows(const float* __restrict__ S, unsigned short* __restrict__ P,
             int n) {
    __shared__ float red[256];
    const int row = blockIdx.x, tid = threadIdx.x;
    const float* s = S + (size_t)row * n;

    float m = -3.402823466e38f;
    for (int c = tid; c < n; c += 256) m = fmaxf(m, s[c]);
    red[tid] = m;
    __syncthreads();
    for (int o = 128; o > 0; o >>= 1) {
        if (tid < o) red[tid] = fmaxf(red[tid], red[tid + o]);
        __syncthreads();
    }
    m = red[0];
    __syncthreads();

    float sum = 0.f;
    for (int c = tid; c < n; c += 256) sum += __expf(s[c] - m);
    red[tid] = sum;
    __syncthreads();
    for (int o = 128; o > 0; o >>= 1) {
        if (tid < o) red[tid] += red[tid + o];
        __syncthreads();
    }
    float inv = 1.0f / red[0];

    unsigned short* p = P + (size_t)row * n;
    for (int c = tid; c < n; c += 256) p[c] = f2bf(__expf(s[c] - m) * inv);
}

// ---------------------------------------------------------------- launch
extern "C" void kernel_launch(void* const* d_in, const int* in_sizes, int n_in,
                              void* d_out, int out_size, void* d_ws,
                              size_t ws_size, hipStream_t stream) {
    const float* x  = (const float*)d_in[0];   // (N, D)
    const float* Wq = (const float*)d_in[1];   // (D, N)
    const float* Wk = (const float*)d_in[2];
    const float* Wv = (const float*)d_in[3];
    float* out = (float*)d_out;                // (N, N)

    const size_t MB = 1024ull * 1024ull;
    char* ws = (char*)d_ws;
    // workspace layout (needs 192 MB):
    unsigned short* xb  = (unsigned short*)(ws + 0);        //  8 MB  x bf16
    unsigned short* wt0 = (unsigned short*)(ws + 8 * MB);   //  8 MB  Wq^T bf16
    unsigned short* wt1 = (unsigned short*)(ws + 16 * MB);
    unsigned short* wt2 = (unsigned short*)(ws + 24 * MB);
    unsigned short* qb  = (unsigned short*)(ws + 32 * MB);  // 32 MB  q bf16
    unsigned short* kb  = (unsigned short*)(ws + 64 * MB);  // 32 MB  k bf16
    unsigned short* vb  = (unsigned short*)(ws + 96 * MB);  // 32 MB  v bf16
    unsigned short* vt  = (unsigned short*)(ws + 0);        // 32 MB  v^T (reuse x/W)
    float*          sc  = (float*)(ws + 128 * MB);          // 64 MB  score f32
    unsigned short* pb  = (unsigned short*)(ws + 32 * MB);  // 32 MB  p (reuse q)

    dim3 tb256(256);

    // 1) x -> bf16
    cvt_f32_bf16<<<4096, tb256, 0, stream>>>(x, xb, NN * DD);

    // 2) W -> W^T bf16 (so GEMM1's B operand is B^T row-major)
    {
        dim3 g(NN / 32, DD / 32), b(32, 8);
        transpose_f32_to_bf16<<<g, b, 0, stream>>>(Wq, wt0, DD, NN);
        transpose_f32_to_bf16<<<g, b, 0, stream>>>(Wk, wt1, DD, NN);
        transpose_f32_to_bf16<<<g, b, 0, stream>>>(Wv, wt2, DD, NN);
    }

    // 3) q,k,v = x @ W   (M=4096, N=4096, K=1024), stored bf16
    {
        dim3 g(NN / BN, NN / BM);
        gemm_bt_bf16<false><<<g, tb256, 0, stream>>>(xb, wt0, qb, NN, NN, DD);
        gemm_bt_bf16<false><<<g, tb256, 0, stream>>>(xb, wt1, kb, NN, NN, DD);
        gemm_bt_bf16<false><<<g, tb256, 0, stream>>>(xb, wt2, vb, NN, NN, DD);
    }

    // 4) v^T for the final GEMM's B operand
    {
        dim3 g(NN / 32, NN / 32), b(32, 8);
        transpose_bf16<<<g, b, 0, stream>>>(vb, vt, NN, NN);
    }

    // 5) score = q @ k^T  -> B^T = k (already row-major). fp32 out.
    {
        dim3 g(NN / BN, NN / BM);
        gemm_bt_bf16<true><<<g, tb256, 0, stream>>>(qb, kb, sc, NN, NN, NN);
    }

    // 6) row softmax, p stored bf16 (overwrites q slot)
    softmax_rows<<<NN, tb256, 0, stream>>>(sc, pb, NN);

    // 7) out = p @ v  -> B^T = v^T. fp32 out to d_out.
    {
        dim3 g(NN / BN, NN / BM);
        gemm_bt_bf16<true><<<g, tb256, 0, stream>>>(pb, vt, out, NN, NN, NN);
    }
}
